// PlayerSelectionNetwork_90546500535091
// MI455X (gfx1250) — matrix operand order repaired and checked by reading, not compiled, not run on hardware
//
#include <hip/hip_runtime.h>

// ---------------------------------------------------------------------------
// Types for CDNA5 WMMA (wave32): v_wmma_f32_16x16x32_bf16
// ---------------------------------------------------------------------------
typedef __attribute__((ext_vector_type(16))) __bf16 v16bf;
typedef __attribute__((ext_vector_type(8)))  __bf16 v8bf;
typedef __attribute__((ext_vector_type(8)))  float  v8f;

union Frag {
    v16bf v;
    v8bf  h[2];
};

__device__ __forceinline__ v8f wmma_bf16(v16bf a, v16bf b, v8f c) {
    // (neg_a, A, neg_b, B, c_mod, C, reuse_a, reuse_b)
    return __builtin_amdgcn_wmma_f32_16x16x32_bf16(false, a, false, b,
                                                   (short)0, c, false, false);
}

__device__ __forceinline__ float sigmoid_(float x) {
    return __fdividef(1.0f, 1.0f + __expf(-x));
}
__device__ __forceinline__ float tanh_(float x) {
    // tanh(x) = 1 - 2/(exp(2x)+1); saturates correctly at +/-inf
    return 1.0f - __fdividef(2.0f, __expf(2.0f * x) + 1.0f);
}

// ---------------------------------------------------------------------------
// Model constants
// ---------------------------------------------------------------------------
#define BATCH  16384
#define T_OBS  10
#define NAG    16
#define IND    4
#define H      64
#define H3     192
#define FEAT   1024   // NAG*H
#define H1DIM  512
#define H2DIM  256
#define NOUT   15

// Workspace layout (bytes)
#define WS_FEATS 0u
#define WS_H1    33554432u              // 16384*1024*2
#define WS_H2    50331648u              // + 16384*512*2
#define WS_W1B   58720256u              // + 16384*256*2
#define WS_W2B   59768832u              // + 1024*512*2
#define WS_WOB   60030976u              // + 512*256*2

// ---------------------------------------------------------------------------
// Kernel 0: convert MLP weights to bf16 (Wout padded 256x15 -> 256x16)
// ---------------------------------------------------------------------------
__global__ __launch_bounds__(256)
void pack_weights_kernel(const float* __restrict__ W1,
                         const float* __restrict__ W2,
                         const float* __restrict__ Wout,
                         __bf16* __restrict__ W1b,
                         __bf16* __restrict__ W2b,
                         __bf16* __restrict__ WoutP) {
    int i = blockIdx.x * 256 + threadIdx.x;
    if (i < FEAT * H1DIM)  W1b[i] = (__bf16)W1[i];
    if (i < H1DIM * H2DIM) W2b[i] = (__bf16)W2[i];
    if (i < H2DIM * 16) {
        int k = i >> 4, c = i & 15;
        WoutP[i] = (c < NOUT) ? (__bf16)Wout[k * NOUT + c] : (__bf16)0.0f;
    }
}

// ---------------------------------------------------------------------------
// Kernel 1: per-agent GRU over T=10 steps.
//   grid = (BATCH/128, NAG), block = 256 (8 wave32; wave w owns rows 16w..16w+15)
//   h@Wh (128x64 @ 64x192) done with v_wmma_f32_16x16x32_bf16, fp32 gates/state.
//   No barriers in the time loop: each wave only touches its own 16 rows.
// ---------------------------------------------------------------------------
__global__ __launch_bounds__(256)
void gru_kernel(const float* __restrict__ x,
                const float* __restrict__ Wi,
                const float* __restrict__ Wh,
                const float* __restrict__ bi,
                const float* __restrict__ bhn,
                __bf16* __restrict__ feats) {
    // Wh pre-swizzled into WMMA B-fragment layout: [j(12)][chunk(2)][lane(32)][i(16)]
    __shared__ __align__(16) __bf16 WhS[12 * 2 * 32 * 16];   // 24576 B
    // h as bf16, row-major [128][64] (wave-private row ranges)
    __shared__ __align__(16) __bf16 hbf[128 * 64];           // 16384 B
    // x tile: [row(128)][t(10)] float4
    __shared__ __align__(16) float4 xs[128 * T_OBS];         // 20480 B

    const int a   = blockIdx.y;
    const int b0  = blockIdx.x * 128;
    const int tid = threadIdx.x;
    const int w   = tid >> 5;     // wave id 0..7
    const int l   = tid & 31;     // lane
    const int lhi = l >> 4;       // 0/1
    const int ll  = l & 15;

    // ---- stage Wh (swizzled bf16) ----
    const float* Wha = Wh + (size_t)a * H * H3;
    for (int e = tid; e < H * H3; e += 256) {
        int k = e / H3, c = e % H3;
        int j = c >> 4, nr = c & 15;
        int chunk = k >> 5, kr = k & 31;
        int lane = nr + (((kr >> 3) & 1) << 4);
        int i = (kr & 7) | ((kr >> 4) << 3);
        WhS[((j * 2 + chunk) * 32 + lane) * 16 + i] = (__bf16)Wha[e];
    }
    // ---- stage x tile ----
    const float* xb = x + (size_t)b0 * (T_OBS * NAG * IND);
    for (int p = tid; p < 128 * T_OBS; p += 256) {
        int r = p & 127, t = p >> 7;
        xs[r * T_OBS + t] =
            *(const float4*)(xb + (size_t)r * (T_OBS * NAG * IND) + t * (NAG * IND) + a * IND);
    }

    // ---- per-lane Wi / bias registers (cols 16j+ll for j=0..3) ----
    float wir[4][4], wiz[4][4], win_[4][4], bir[4], biz[4], bin_[4], bhnr[4];
    {
        const float* Wia = Wi + (size_t)a * IND * H3;
        const float* bia = bi + (size_t)a * H3;
        const float* bha = bhn + (size_t)a * H;
#pragma unroll
        for (int j = 0; j < 4; j++) {
            int c = j * 16 + ll;
#pragma unroll
            for (int d = 0; d < 4; d++) {
                wir[j][d]  = Wia[d * H3 + c];
                wiz[j][d]  = Wia[d * H3 + 64 + c];
                win_[j][d] = Wia[d * H3 + 128 + c];
            }
            bir[j] = bia[c]; biz[j] = bia[64 + c]; bin_[j] = bia[128 + c];
            bhnr[j] = bha[c];
        }
    }

    float hold[8][4];  // fp32 state: [v][j] -> h[row = 16w+v+8*lhi][col = 16j+ll]
#pragma unroll
    for (int v = 0; v < 8; v++)
#pragma unroll
        for (int j = 0; j < 4; j++) hold[v][j] = 0.0f;

    __syncthreads();

#pragma unroll 1
    for (int t = 0; t < T_OBS; t++) {
        Frag a0, a1;
        if (t > 0) {
            const __bf16* hrow = hbf + (w * 16 + ll) * 64;
            a0.h[0] = *(const v8bf*)(hrow + lhi * 8);
            a0.h[1] = *(const v8bf*)(hrow + 16 + lhi * 8);
            a1.h[0] = *(const v8bf*)(hrow + 32 + lhi * 8);
            a1.h[1] = *(const v8bf*)(hrow + 48 + lhi * 8);
        }
#pragma unroll
        for (int j = 0; j < 4; j++) {
            v8f ar = {}, az = {}, an = {};
            if (t > 0) {
                Frag br0, br1, bz0, bz1, bn0, bn1;
                const __bf16* p;
                p = WhS + (((j    ) * 2 + 0) * 32 + l) * 16; br0.h[0] = *(const v8bf*)p; br0.h[1] = *(const v8bf*)(p + 8);
                p = WhS + (((j    ) * 2 + 1) * 32 + l) * 16; br1.h[0] = *(const v8bf*)p; br1.h[1] = *(const v8bf*)(p + 8);
                p = WhS + (((j + 4) * 2 + 0) * 32 + l) * 16; bz0.h[0] = *(const v8bf*)p; bz0.h[1] = *(const v8bf*)(p + 8);
                p = WhS + (((j + 4) * 2 + 1) * 32 + l) * 16; bz1.h[0] = *(const v8bf*)p; bz1.h[1] = *(const v8bf*)(p + 8);
                p = WhS + (((j + 8) * 2 + 0) * 32 + l) * 16; bn0.h[0] = *(const v8bf*)p; bn0.h[1] = *(const v8bf*)(p + 8);
                p = WhS + (((j + 8) * 2 + 1) * 32 + l) * 16; bn1.h[0] = *(const v8bf*)p; bn1.h[1] = *(const v8bf*)(p + 8);
                ar = wmma_bf16(a0.v, br0.v, ar); ar = wmma_bf16(a1.v, br1.v, ar);
                az = wmma_bf16(a0.v, bz0.v, az); az = wmma_bf16(a1.v, bz1.v, az);
                an = wmma_bf16(a0.v, bn0.v, an); an = wmma_bf16(a1.v, bn1.v, an);
            }
#pragma unroll
            for (int v = 0; v < 8; v++) {
                int rl = w * 16 + v + 8 * lhi;       // row within 128-tile
                float4 xv = xs[rl * T_OBS + t];
                float xr = bir[j]  + xv.x * wir[j][0]  + xv.y * wir[j][1]  + xv.z * wir[j][2]  + xv.w * wir[j][3];
                float xz = biz[j]  + xv.x * wiz[j][0]  + xv.y * wiz[j][1]  + xv.z * wiz[j][2]  + xv.w * wiz[j][3];
                float xn = bin_[j] + xv.x * win_[j][0] + xv.y * win_[j][1] + xv.z * win_[j][2] + xv.w * win_[j][3];
                float r = sigmoid_(xr + ar[v]);
                float z = sigmoid_(xz + az[v]);
                float n = tanh_(xn + r * (an[v] + bhnr[j]));
                float hn2 = (1.0f - z) * n + z * hold[v][j];
                hold[v][j] = hn2;
                hbf[rl * 64 + j * 16 + ll] = (__bf16)hn2;   // wave-private rows
            }
        }
    }

    // write features (bf16) for MLP: feats[b][a*64 + hcol]
#pragma unroll
    for (int v = 0; v < 8; v++) {
        size_t rg = (size_t)(b0 + w * 16 + v + 8 * lhi);
#pragma unroll
        for (int j = 0; j < 4; j++)
            feats[rg * FEAT + a * H + j * 16 + ll] = (__bf16)hold[v][j];
    }
}

// ---------------------------------------------------------------------------
// Kernel 2: WMMA GEMM + bias + activation, K=64 double-buffered LDS pipeline.
//   Tile: M=128, N=NTILES*16. One barrier per 64-K round; 2*NTILES WMMA/wave.
//   ACT 0 = relu (bf16 out, width N), ACT 1 = sigmoid (fp32 out, width NReal)
// ---------------------------------------------------------------------------
template <int NTILES, int ACT>
__global__ __launch_bounds__(256)
void gemm_bias_act_kernel(const __bf16* __restrict__ A,
                          const __bf16* __restrict__ W,
                          const float* __restrict__ bias,
                          void* __restrict__ Out,
                          int N, int K, int NReal) {
    // ping-pong buffers
    __shared__ __align__(16) __bf16 As[2 * 128 * 64];               // 32 KB
    __shared__ __align__(16) __bf16 Bs[2 * NTILES * 2 * 32 * 16];   // <= 32 KB

    const int tid = threadIdx.x;
    const int w   = tid >> 5;
    const int l   = tid & 31;
    const int lhi = l >> 4;
    const int ll  = l & 15;
    const int m0  = blockIdx.x * 128;
    const int n0  = blockIdx.y * (NTILES * 16);
    const int NT16 = NTILES * 16;

    v8f acc[NTILES];
#pragma unroll
    for (int j = 0; j < NTILES; j++) acc[j] = (v8f){};

    // ---- stage one 64-wide K chunk into buffer b ----
    auto stage = [&](int k0, int b) {
        __bf16* Ab = As + b * (128 * 64);
        __bf16* Bb = Bs + b * (NTILES * 2 * 32 * 16);
        // A chunk [128][64] row-major, b128 global loads
        for (int g = tid; g < 128 * 8; g += 256) {
            int row = g >> 3, kk8 = (g & 7) * 8;
            *(v8bf*)(Ab + row * 64 + kk8) =
                *(const v8bf*)(A + (size_t)(m0 + row) * K + k0 + kk8);
        }
        // B chunk [64][NT16] swizzled into fragment layout [kc][j][lane][i]
        for (int e = tid; e < 64 * NT16; e += 256) {
            int kr = e / NT16, nr = e % NT16;
            int kc = kr >> 5, krr = kr & 31;
            int j = nr >> 4;
            int lane = (nr & 15) + (((krr >> 3) & 1) << 4);
            int i = (krr & 7) | ((krr >> 4) << 3);
            Bb[((kc * NTILES + j) * 32 + lane) * 16 + i] =
                W[(size_t)(k0 + kr) * N + n0 + nr];
        }
        // prefetch the chunk after this one (global_prefetch_b8)
        int kp = k0 + 64;
        if (kp < K) {
            __builtin_prefetch(A + (size_t)(m0 + (tid & 127)) * K + kp, 0, 3);
            if (tid < 64)
                __builtin_prefetch(W + (size_t)(kp + tid) * N + n0, 0, 3);
        }
    };

    // ---- consume buffer b: 2*NTILES WMMAs per wave ----
    auto compute = [&](int b) {
        const __bf16* Ab = As + b * (128 * 64);
        const __bf16* Bb = Bs + b * (NTILES * 2 * 32 * 16);
        Frag aF0, aF1;
        const __bf16* arow = Ab + (w * 16 + ll) * 64;
        aF0.h[0] = *(const v8bf*)(arow + lhi * 8);
        aF0.h[1] = *(const v8bf*)(arow + 16 + lhi * 8);
        aF1.h[0] = *(const v8bf*)(arow + 32 + lhi * 8);
        aF1.h[1] = *(const v8bf*)(arow + 48 + lhi * 8);
#pragma unroll
        for (int j = 0; j < NTILES; j++) {
            Frag b0, b1;
            const __bf16* bp0 = Bb + ((0 * NTILES + j) * 32 + l) * 16;
            const __bf16* bp1 = Bb + ((1 * NTILES + j) * 32 + l) * 16;
            b0.h[0] = *(const v8bf*)bp0; b0.h[1] = *(const v8bf*)(bp0 + 8);
            b1.h[0] = *(const v8bf*)bp1; b1.h[1] = *(const v8bf*)(bp1 + 8);
            acc[j] = wmma_bf16(aF0.v, b0.v, acc[j]);
            acc[j] = wmma_bf16(aF1.v, b1.v, acc[j]);
        }
    };

    // ---- software pipeline over K ----
    stage(0, 0);
    int buf = 0;
#pragma unroll 1
    for (int k0 = 64; k0 < K; k0 += 64) {
        __syncthreads();          // buf fully staged; buf^1 no longer being read
        stage(k0, buf ^ 1);       // overlap next chunk's global loads with WMMA
        compute(buf);
        buf ^= 1;
    }
    __syncthreads();
    compute(buf);

    // ---- epilogue: bias + activation + store ----
#pragma unroll
    for (int j = 0; j < NTILES; j++) {
        int col = n0 + j * 16 + ll;
        float bv = (ACT == 1) ? ((col < NReal) ? bias[col] : 0.0f) : bias[col];
#pragma unroll
        for (int v = 0; v < 8; v++) {
            int row = m0 + w * 16 + v + 8 * lhi;
            float val = acc[j][v] + bv;
            if (ACT == 1) {
                if (col < NReal)
                    ((float*)Out)[(size_t)row * NReal + col] = sigmoid_(val);
            } else {
                ((__bf16*)Out)[(size_t)row * N + col] = (__bf16)fmaxf(val, 0.0f);
            }
        }
    }
}

// ---------------------------------------------------------------------------
// Launcher
// ---------------------------------------------------------------------------
extern "C" void kernel_launch(void* const* d_in, const int* in_sizes, int n_in,
                              void* d_out, int out_size, void* d_ws, size_t ws_size,
                              hipStream_t stream) {
    const float* x    = (const float*)d_in[0];
    const float* Wi   = (const float*)d_in[1];
    const float* Wh   = (const float*)d_in[2];
    const float* bi   = (const float*)d_in[3];
    const float* bhn  = (const float*)d_in[4];
    const float* W1   = (const float*)d_in[5];
    const float* b1   = (const float*)d_in[6];
    const float* W2   = (const float*)d_in[7];
    const float* b2   = (const float*)d_in[8];
    const float* Wout = (const float*)d_in[9];
    const float* bout = (const float*)d_in[10];

    char* ws = (char*)d_ws;
    __bf16* feats = (__bf16*)(ws + WS_FEATS);
    __bf16* h1    = (__bf16*)(ws + WS_H1);
    __bf16* h2    = (__bf16*)(ws + WS_H2);
    __bf16* W1b   = (__bf16*)(ws + WS_W1B);
    __bf16* W2b   = (__bf16*)(ws + WS_W2B);
    __bf16* WoutP = (__bf16*)(ws + WS_WOB);

    // 0) pack MLP weights to bf16
    pack_weights_kernel<<<(FEAT * H1DIM + 255) / 256, 256, 0, stream>>>(
        W1, W2, Wout, W1b, W2b, WoutP);

    // 1) GRU over 10 steps -> feats[B][1024] bf16
    gru_kernel<<<dim3(BATCH / 128, NAG), 256, 0, stream>>>(
        x, Wi, Wh, bi, bhn, feats);

    // 2) h1 = relu(feats @ W1 + b1)        (16384x1024 @ 1024x512)
    gemm_bias_act_kernel<8, 0><<<dim3(BATCH / 128, H1DIM / 128), 256, 0, stream>>>(
        feats, W1b, b1, (void*)h1, H1DIM, FEAT, H1DIM);

    // 3) h2 = relu(h1 @ W2 + b2)           (16384x512 @ 512x256)
    gemm_bias_act_kernel<8, 0><<<dim3(BATCH / 128, H2DIM / 128), 256, 0, stream>>>(
        h1, W2b, b2, (void*)h2, H2DIM, H1DIM, H2DIM);

    // 4) out = sigmoid(h2 @ Wout + bout)   (16384x256 @ 256x16->15, fp32 out)
    gemm_bias_act_kernel<1, 1><<<dim3(BATCH / 128, 1), 256, 0, stream>>>(
        h2, WoutP, bout, d_out, 16, H2DIM, NOUT);
}
